// Attention_36172214567562
// MI455X (gfx1250) — matrix operand order repaired and checked by reading, not compiled, not run on hardware
//
#include <hip/hip_runtime.h>
#include <cstdint>

typedef __attribute__((ext_vector_type(16))) __bf16 v16bf;
typedef __attribute__((ext_vector_type(8)))  float  v8f;
typedef __attribute__((ext_vector_type(4)))  float  f4;

#define BATCH   64
#define RDIM    2048
#define DV      1024
#define DQ      1024
#define NH      512

#define BS_STRIDE  40            // 32 K + 8 pad (bank spread)
#define BS_ELEMS   (256 * BS_STRIDE)

// ---------------- Kernel 1: qp = q @ Wy + by  -> ws.qp [64,512] ----------------
__global__ __launch_bounds__(256) void qp_kernel(const float* __restrict__ q,
                                                 const float* __restrict__ Wy,
                                                 const float* __restrict__ by,
                                                 float* __restrict__ qp) {
    int idx = blockIdx.x * 256 + threadIdx.x;   // 0..32767
    int b = idx >> 9;
    int n = idx & 511;
    const float* qr = q + (size_t)b * DQ;
    const float* wc = Wy + n;
    float acc = by[n];
    for (int k = 0; k < DQ; ++k) acc += qr[k] * wc[(size_t)k * NH];
    qp[idx] = acc;
}

// ------------- Kernel 2: WxT[n,k] = bf16(Wx[k,n])  -> ws.wxt [512,1024] -------------
__global__ __launch_bounds__(256) void wxt_kernel(const float* __restrict__ Wx,
                                                  __bf16* __restrict__ wxt) {
    int idx = blockIdx.x * 256 + threadIdx.x;   // 0..524287
    int n = idx >> 10;
    int k = idx & 1023;
    wxt[idx] = (__bf16)Wx[(size_t)k * NH + n];
}

// ---------------- helpers for the WMMA kernel ----------------
struct AFrag { f4 x0, x1, x2, x3; };

__device__ __forceinline__ AFrag load_af(const float* p) {
    AFrag r;
    r.x0 = *(const f4*)(p);
    r.x1 = *(const f4*)(p + 4);
    r.x2 = *(const f4*)(p + 16);
    r.x3 = *(const f4*)(p + 20);
    return r;
}

__device__ __forceinline__ v16bf cvt_a(const AFrag& af) {
    v16bf a;
    #pragma unroll
    for (int i = 0; i < 4; ++i) {
        a[i]      = (__bf16)af.x0[i];
        a[4 + i]  = (__bf16)af.x1[i];
        a[8 + i]  = (__bf16)af.x2[i];
        a[12 + i] = (__bf16)af.x3[i];
    }
    return a;
}

// ------- Kernel 3: fused h=tanh(v@Wx + bx + qp), logits = h@Wa  (WMMA bf16) -------
// grid = 64 batches * 16 row-tiles = 1024 blocks, 256 threads (8 waves of 16 rows)
// B tile double-buffered in LDS; A + B-next loads issued before the WMMA stream.
__global__ __launch_bounds__(256) void gemm_logits_kernel(
    const float*  __restrict__ v,
    const float*  __restrict__ bx,
    const float*  __restrict__ wa,
    const float*  __restrict__ qp,
    const __bf16* __restrict__ wxt,
    float*        __restrict__ logits) {

    __shared__ __align__(16) __bf16 Bs[2 * BS_ELEMS];  // 2 x (256 N rows x 32 K, stride 40)
    __shared__ float sbias[NH];
    __shared__ float wal[NH];

    const int bid   = blockIdx.x;
    const int b     = bid >> 4;
    const int rtile = bid & 15;
    const int tid   = threadIdx.x;
    const int wm    = tid >> 5;      // wave id: its 16-row subtile
    const int lane  = tid & 31;
    const int lm    = lane & 15;
    const int kh    = lane >> 4;     // half-select

    for (int i = tid; i < NH; i += 256) {
        sbias[i] = bx[i] + qp[b * NH + i];
        wal[i]   = wa[i];
    }

    const int row = rtile * 128 + wm * 16 + lm;              // A: M = lane&15
    const float* arow = v + ((size_t)b * RDIM + row) * DV;

    float lacc[8];
    #pragma unroll
    for (int r = 0; r < 8; ++r) lacc[r] = 0.0f;

    for (int pass = 0; pass < 2; ++pass) {
        const int nbase = pass * 256;
        const __bf16* brow = wxt + (size_t)(nbase + tid) * DV;

        v8f c[16];
        #pragma unroll
        for (int t = 0; t < 16; ++t)
            #pragma unroll
            for (int r = 0; r < 8; ++r) c[t][r] = 0.0f;

        // ---- prologue: stage B tile 0 into buffer 0, preload A fragment 0 ----
        {
            const uint4* src = (const uint4*)(brow);
            uint4* dst = (uint4*)(Bs + tid * BS_STRIDE);
            dst[0] = src[0]; dst[1] = src[1]; dst[2] = src[2]; dst[3] = src[3];
        }
        AFrag af_cur = load_af(arow + kh * 8);
        __syncthreads();

        for (int it = 0; it < 32; ++it) {
            const int cur = it & 1;
            const bool has_next = (it + 1) < 32;

            // issue next-tile loads early: latency hidden under the WMMA stream
            uint4 nb0, nb1, nb2, nb3;
            AFrag af_nxt;
            if (has_next) {
                const int k0n = (it + 1) * 32;
                const uint4* src = (const uint4*)(brow + k0n);
                nb0 = src[0]; nb1 = src[1]; nb2 = src[2]; nb3 = src[3];
                af_nxt = load_af(arow + k0n + kh * 8);
            }

            // A fragment (16x32 bf16): e<8 -> K = kh*8+e ; e>=8 -> K = 16+kh*8+(e-8)
            const v16bf a = cvt_a(af_cur);

            const __bf16* bbase = Bs + cur * BS_ELEMS + lm * BS_STRIDE + kh * 8;
            #pragma unroll
            for (int t = 0; t < 16; ++t) {
                // B fragment (32x16 bf16): lane = N, e<8 -> K = kh*8+e ; e>=8 -> K = 16+kh*8+(e-8)
                union { uint4 u[2]; v16bf vv; } bb;
                const uint4* bp = (const uint4*)(bbase + t * 16 * BS_STRIDE);
                bb.u[0] = bp[0];
                bb.u[1] = bp[2];
                c[t] = __builtin_amdgcn_wmma_f32_16x16x32_bf16(
                    false, a, false, bb.vv, (short)0, c[t], false, false);
            }

            if (has_next) {
                uint4* dst = (uint4*)(Bs + (cur ^ 1) * BS_ELEMS + tid * BS_STRIDE);
                dst[0] = nb0; dst[1] = nb1; dst[2] = nb2; dst[3] = nb3;
                af_cur = af_nxt;
            }
            __syncthreads();
        }

        // epilogue: h = tanh(c + bx + qp); lacc += h * Wa
        #pragma unroll
        for (int t = 0; t < 16; ++t) {
            const int n = nbase + t * 16 + lm;
            const float sb = sbias[n];
            const float w  = wal[n];
            #pragma unroll
            for (int r = 0; r < 8; ++r) {
                float h = tanhf(c[t][r] + sb);
                lacc[r] += h * w;
            }
        }
        __syncthreads();   // all waves done with pass (Bs reuse safety)
    }

    // reduce over the 16 N-lanes (keep within half: masks 8,4,2,1)
    #pragma unroll
    for (int r = 0; r < 8; ++r) {
        float s = lacc[r];
        s += __shfl_xor(s, 8, 32);
        s += __shfl_xor(s, 4, 32);
        s += __shfl_xor(s, 2, 32);
        s += __shfl_xor(s, 1, 32);
        lacc[r] = s;
    }
    if (lm == 0) {
        const int rowbase = rtile * 128 + wm * 16 + kh * 8;   // C/D: M = r + 8*half
        #pragma unroll
        for (int r = 0; r < 8; ++r)
            logits[(size_t)b * RDIM + rowbase + r] = lacc[r];
    }
}

// ---------------- Kernel 4: softmax over R, in place in d_out ----------------
__global__ __launch_bounds__(256) void softmax_kernel(float* __restrict__ logits) {
    __shared__ float red[256];
    const int b = blockIdx.x;
    const int tid = threadIdx.x;
    float* lp = logits + (size_t)b * RDIM;

    float x[8];
    float m = -3.4e38f;
    #pragma unroll
    for (int i = 0; i < 8; ++i) { x[i] = lp[tid + i * 256]; m = fmaxf(m, x[i]); }
    red[tid] = m; __syncthreads();
    for (int s = 128; s > 0; s >>= 1) {
        if (tid < s) red[tid] = fmaxf(red[tid], red[tid + s]);
        __syncthreads();
    }
    m = red[0]; __syncthreads();

    float e[8]; float sum = 0.0f;
    #pragma unroll
    for (int i = 0; i < 8; ++i) { e[i] = __expf(x[i] - m); sum += e[i]; }
    red[tid] = sum; __syncthreads();
    for (int s = 128; s > 0; s >>= 1) {
        if (tid < s) red[tid] += red[tid + s];
        __syncthreads();
    }
    const float inv = 1.0f / red[0];
    #pragma unroll
    for (int i = 0; i < 8; ++i) lp[tid + i * 256] = e[i] * inv;
}

// ------------- Kernel 5: partial weighted sums over 128-row segments -------------
// v is stream-once here: use non-temporal loads to avoid polluting L2.
__global__ __launch_bounds__(256) void wsum_partial_kernel(const float* __restrict__ v,
                                                           const float* __restrict__ probs,
                                                           float* __restrict__ partials) {
    const int bid = blockIdx.x;       // 1024 = 64 b * 16 segs
    const int b = bid >> 4;
    const int seg = bid & 15;
    const int d = threadIdx.x * 4;
    const float* vb = v + ((size_t)b * RDIM + seg * 128) * DV;
    const float* pb = probs + (size_t)b * RDIM + seg * 128;
    f4 acc = {0.f, 0.f, 0.f, 0.f};
    for (int r = 0; r < 128; ++r) {
        const float p = pb[r];
        const f4 vv = __builtin_nontemporal_load((const f4*)(vb + (size_t)r * DV + d));
        acc += p * vv;
    }
    *(f4*)(partials + (size_t)bid * DV + d) = acc;
}

// ---------------- Kernel 6: reduce 16 partials -> v_weighted ----------------
__global__ __launch_bounds__(256) void wsum_reduce_kernel(const float* __restrict__ partials,
                                                          float* __restrict__ out) {
    const int b = blockIdx.x;
    const int d = threadIdx.x * 4;
    f4 acc = {0.f, 0.f, 0.f, 0.f};
    for (int seg = 0; seg < 16; ++seg) {
        acc += *(const f4*)(partials + ((size_t)(b * 16 + seg)) * DV + d);
    }
    *(f4*)(out + (size_t)b * DV + d) = acc;
}

extern "C" void kernel_launch(void* const* d_in, const int* in_sizes, int n_in,
                              void* d_out, int out_size, void* d_ws, size_t ws_size,
                              hipStream_t stream) {
    const float* v  = (const float*)d_in[0];
    const float* q  = (const float*)d_in[1];
    const float* Wx = (const float*)d_in[2];
    const float* bx = (const float*)d_in[3];
    const float* Wy = (const float*)d_in[4];
    const float* by = (const float*)d_in[5];
    const float* Wa = (const float*)d_in[6];
    // ba (d_in[7]) is softmax-shift-invariant: both outputs independent of it.

    float* vw     = (float*)d_out;                 // [64,1024]
    float* probs  = (float*)d_out + BATCH * DV;    // [64,2048] (logits first, then in-place softmax)

    uint8_t* ws   = (uint8_t*)d_ws;
    float*  qp       = (float*)(ws);                         // 32768 f  (128 KB)
    __bf16* wxt      = (__bf16*)(ws + 131072);               // 524288 bf16 (1 MB)
    float*  partials = (float*)(ws + 131072 + 1048576);      // 1024*1024 f (4 MB)

    qp_kernel          <<<128,  256, 0, stream>>>(q, Wy, by, qp);
    wxt_kernel         <<<2048, 256, 0, stream>>>(Wx, wxt);
    gemm_logits_kernel <<<1024, 256, 0, stream>>>(v, bx, Wa, qp, wxt, probs);
    softmax_kernel     <<<64,   256, 0, stream>>>(probs);
    wsum_partial_kernel<<<1024, 256, 0, stream>>>(v, probs, partials);
    wsum_reduce_kernel <<<64,   256, 0, stream>>>(partials, vw);
}
